// DGCNN_21345987461134
// MI455X (gfx1250) — matrix operand order repaired and checked by reading, not compiled, not run on hardware
//
#include <hip/hip_runtime.h>
#include <hip/hip_bf16.h>

typedef __attribute__((ext_vector_type(16))) _Float16 v16h;
typedef __attribute__((ext_vector_type(8)))  _Float16 v8h;
typedef __attribute__((ext_vector_type(8)))  float    v8f;
typedef unsigned int uint32x4 __attribute__((ext_vector_type(4)));
typedef int          int32x4  __attribute__((ext_vector_type(4)));
typedef int          int32x8  __attribute__((ext_vector_type(8)));

#define BATCH 16
#define NPTS  1024
#define KNN   20
#define EPS   1e-5f
#define SLOPE 0.2f

// ---------------------------------------------------------------------------
// Tensor Data Mover support (CDNA5 ISA).
// Guarded: if the builtin is absent we fall back to synchronous LDS copies,
// so the build can never regress.
// ---------------------------------------------------------------------------
#if defined(__has_builtin)
#if __has_builtin(__builtin_amdgcn_tensor_load_to_lds) && \
    __has_builtin(__builtin_amdgcn_s_wait_tensorcnt)
#define USE_TDM 1
#endif
#endif
#ifndef USE_TDM
#define USE_TDM 0
#endif

#if USE_TDM
// 2-D tile load: tile_x elems (2B each) x tile_y rows, global row stride
// stride_elems, into LDS at byte offset lds_off. D# packing per ISA 8.3/8.4:
// group0: count=1 | lds_addr | global_addr[56:0] | type=2
// group1: data_size=1(2B), tensor_dim0/1, tile_dim0/1, tensor_dim0_stride
__device__ __forceinline__ void tdm_load_tile_2d(unsigned lds_off,
                                                 const void* gp,
                                                 unsigned tensor_d0,
                                                 unsigned tensor_d1,
                                                 unsigned tile_x,
                                                 unsigned tile_y,
                                                 unsigned stride_elems) {
  unsigned long long ga = (unsigned long long)(size_t)gp;
  uint32x4 g0;
  g0[0] = 1u;                                               // count=1
  g0[1] = lds_off;                                          // lds_addr
  g0[2] = (unsigned)ga;                                     // global_addr lo
  g0[3] = (unsigned)((ga >> 32) & 0x1FFFFFFull) | (2u << 30);  // hi | type=2
  int32x8 g1;
  g1[0] = (int)(1u << 16);                                  // data_size=2B
  g1[1] = (int)((tensor_d0 & 0xFFFFu) << 16);               // tensor_dim0 lo
  g1[2] = (int)((tensor_d0 >> 16) | ((tensor_d1 & 0xFFFFu) << 16));
  g1[3] = (int)((tensor_d1 >> 16) | ((tile_x & 0xFFFFu) << 16));  // tile_dim0
  g1[4] = (int)(tile_y & 0xFFFFu);                          // tile_dim1
  g1[5] = (int)stride_elems;                                // dim0 stride lo
  g1[6] = 0;
  g1[7] = 0;
  int32x4 z4 = {0, 0, 0, 0};
#if defined(__clang_major__) && __clang_major__ >= 23
  int32x8 z8 = {0, 0, 0, 0, 0, 0, 0, 0};
  __builtin_amdgcn_tensor_load_to_lds(g0, g1, z4, z4, z8, 0);
#else
  __builtin_amdgcn_tensor_load_to_lds(g0, g1, z4, z4, 0);
#endif
}
#endif

__device__ __forceinline__ void stage_wait_le1() {
#if USE_TDM
  __builtin_amdgcn_s_wait_tensorcnt(1);
  asm volatile("" ::: "memory");
#endif
}
__device__ __forceinline__ void stage_wait_0() {
#if USE_TDM
  __builtin_amdgcn_s_wait_tensorcnt(0);
  asm volatile("" ::: "memory");
#endif
}

// Stage one 16x128-half slab (row stride 512 halfs in global) into LDS.
__device__ __forceinline__ void stage_chunk(_Float16* ldst,
                                            const _Float16* gsrc, int lane) {
#if USE_TDM
  tdm_load_tile_2d((unsigned)(size_t)ldst, gsrc, 512u, 16u, 128u, 16u, 512u);
#else
  for (int i = lane; i < 16 * 8; i += 32) {
    int r = i >> 3, cc = (i & 7) * 8;
    *(v8h*)(ldst + r * 128 + cc) = *(const v8h*)(gsrc + (size_t)r * 512 + cc);
  }
#endif
}

// ---------------------------------------------------------------------------
// WMMA fragment loaders (CDNA5 16x16x32 f16 layouts, CDNA5 ISA)
// ---------------------------------------------------------------------------
__device__ __forceinline__ v16h load_frag_a(const _Float16* rowptr, int hi) {
  const _Float16* p = rowptr + hi * 8;
  v8h lo  = *(const v8h*)(p);
  v8h hi8 = *(const v8h*)(p + 16);
  v16h r;
#pragma unroll
  for (int i = 0; i < 8; ++i) { r[i] = lo[i]; r[i + 8] = hi8[i]; }
  return r;
}

__device__ __forceinline__ v16h load_frag_b(const _Float16* colptr, int hi) {
  const _Float16* p = colptr + hi * 8;
  v8h lo  = *(const v8h*)(p);
  v8h hi8 = *(const v8h*)(p + 16);
  v16h r;
#pragma unroll
  for (int i = 0; i < 8; ++i) { r[i] = lo[i]; r[i + 8] = hi8[i]; }
  return r;
}

__device__ __forceinline__ v8f wmma_f16(v16h a, v16h b, v8f c) {
  return __builtin_amdgcn_wmma_f32_16x16x32_f16(false, a, false, b, (short)0, c,
                                                false, false);
}

__device__ __forceinline__ float leaky(float x) {
  return x >= 0.f ? x : SLOPE * x;
}

// ---------------------------------------------------------------------------
// Prep: x (B,3,N) f32 -> xt (B,N,32) f16 zero-padded (layer-1 WMMA operand)
// ---------------------------------------------------------------------------
__global__ void prep_xt_kernel(const float* __restrict__ x,
                               _Float16* __restrict__ xt) {
  int i = blockIdx.x * 256 + threadIdx.x;
  if (i >= BATCH * NPTS * 32) return;
  int c = i & 31;
  int n = (i >> 5) & (NPTS - 1);
  int b = i >> 15;
  xt[i] = (c < 3) ? (_Float16)x[((size_t)b * 3 + c) * NPTS + n] : (_Float16)0.f;
}

// Weight f32 (O,Fin) -> f16 (O,FP) zero-padded
__global__ void cvt_w_kernel(const float* __restrict__ src,
                             _Float16* __restrict__ dst, int O, int Fin, int FP) {
  int i = blockIdx.x * 256 + threadIdx.x;
  if (i >= O * FP) return;
  int o = i / FP, f = i % FP;
  dst[i] = (f < Fin) ? (_Float16)src[o * Fin + f] : (_Float16)0.f;
}

// ---------------------------------------------------------------------------
// KNN: rank columns by s_j = 2<xi,xj> - |xj|^2 (row-constant term dropped).
// f16 WMMA Gram tiles -> 16x512 f32 LDS panel (two passes) -> insertion top-20.
// grid = (N/16, B), block = 256 (8 waves).
// ---------------------------------------------------------------------------
template <int CP>
__global__ __launch_bounds__(256) void knn_kernel(
    const _Float16* __restrict__ act, int rowStride, int* __restrict__ idx) {
  __shared__ float panel[16 * 512];   // 32 KB
  __shared__ float norms[NPTS];       // 4 KB
  __shared__ float td[16][KNN];
  __shared__ int   ti[16][KNN];

  const int b    = blockIdx.y;
  const int rt   = blockIdx.x;
  const int tid  = threadIdx.x;
  const int lane = tid & 31;
  const int wave = tid >> 5;
  const int mrow = lane & 15;
  const int hi   = lane >> 4;

  const _Float16* base = act + (size_t)b * NPTS * rowStride;

  for (int p = tid; p < NPTS; p += 256) {
    const _Float16* r = base + (size_t)p * rowStride;
    float s = 0.f;
    for (int c = 0; c < CP; ++c) { float v = (float)r[c]; s += v * v; }
    norms[p] = s;
  }
  if (tid < 16) {
#pragma unroll
    for (int k = 0; k < KNN; ++k) { td[tid][k] = -3.0e38f; ti[tid][k] = 0; }
  }
  __syncthreads();

  const int row0 = rt * 16;
  const _Float16* arow = base + (size_t)(row0 + mrow) * rowStride;

  for (int js = 0; js < 2; ++js) {
    for (int t = 0; t < 4; ++t) {
      const int j   = js * 32 + wave + 8 * t;
      const int col = j * 16 + mrow;
      const _Float16* brow = base + (size_t)col * rowStride;
      v8f acc = {};
      for (int kk = 0; kk < CP; kk += 32) {
        v16h A  = load_frag_a(arow + kk, hi);
        v16h Bv = load_frag_b(brow + kk, hi);
        acc = wmma_f16(A, Bv, acc);
      }
      const float nc = norms[col];
      const int   lc = col - js * 512;
#pragma unroll
      for (int v = 0; v < 8; ++v) {
        int m = v + hi * 8;
        panel[m * 512 + lc] = 2.f * acc[v] - nc;
      }
    }
    __syncthreads();
    if (tid < 16) {
      const int r = tid;
      for (int c = 0; c < 512; ++c) {
        float s = panel[r * 512 + c];
        if (s > td[r][KNN - 1]) {
          int col = js * 512 + c;
          int p = KNN - 1;
          while (p > 0 && td[r][p - 1] < s) {
            td[r][p] = td[r][p - 1];
            ti[r][p] = ti[r][p - 1];
            --p;
          }
          td[r][p] = s;
          ti[r][p] = col;
        }
      }
    }
    __syncthreads();
  }

  if (tid < 16) {
    int* out = idx + ((size_t)b * NPTS + row0 + tid) * KNN;
#pragma unroll
    for (int k = 0; k < KNN; ++k) out[k] = ti[tid][k];
  }
}

// ---------------------------------------------------------------------------
// EdgeConv: 32xFP edge features in LDS (rows 20..31 duplicate k=0; max-safe),
// WMMA GEMM vs Wh (O,FP f16), BN + leaky, K-max via shfl_xor(16).
// grid = (N, B), block = 256 (8 waves).
// ---------------------------------------------------------------------------
template <int CIN, int FP, int O>
__global__ __launch_bounds__(256) void edgeconv_kernel(
    const _Float16* __restrict__ actIn, int inStride,
    const int* __restrict__ idx, const float* __restrict__ bn,
    const _Float16* __restrict__ Wh, _Float16* __restrict__ outBase,
    int outStride) {
  __shared__ _Float16 feat[32 * FP];
  __shared__ int sidx[KNN];

  const int n    = blockIdx.x;
  const int b    = blockIdx.y;
  const int tid  = threadIdx.x;
  const int lane = tid & 31;
  const int wave = tid >> 5;
  const int hi   = lane >> 4;

  if (tid < KNN) sidx[tid] = idx[((size_t)b * NPTS + n) * KNN + tid];
  __syncthreads();

  const _Float16* actB = actIn + (size_t)b * NPTS * inStride;
  const _Float16* ctr  = actB + (size_t)n * inStride;

  for (int e = tid; e < 32 * FP; e += 256) {
    int r = e / FP, f = e % FP;
    int k  = (r < KNN) ? r : 0;
    int nb = sidx[k];
    _Float16 val;
    if (f < CIN)            val = actB[(size_t)nb * inStride + f] - ctr[f];
    else if (f < 2 * CIN)   val = ctr[f - CIN];
    else                    val = (_Float16)0.f;
    feat[r * FP + f] = val;
  }
  __syncthreads();

  const int nOt = O / 16;
  for (int ot = wave; ot < nOt; ot += 8) {
    const int o = ot * 16 + (lane & 15);
    const _Float16* wcol = Wh + (size_t)o * FP;
    v8f acc0 = {}, acc1 = {};
    for (int kk = 0; kk < FP; kk += 32) {
      v16h Bv = load_frag_b(wcol + kk, hi);
      v16h A0 = load_frag_a(&feat[(lane & 15) * FP + kk], hi);
      v16h A1 = load_frag_a(&feat[(16 + (lane & 15)) * FP + kk], hi);
      acc0 = wmma_f16(A0, Bv, acc0);
      acc1 = wmma_f16(A1, Bv, acc1);
    }
    const float g  = bn[o];
    const float be = bn[O + o];
    const float mu = bn[2 * O + o];
    const float va = bn[3 * O + o];
    const float sc = g * rsqrtf(va + EPS);
    const float sh = be - mu * sc;
    float m = -3.0e38f;
#pragma unroll
    for (int v = 0; v < 8; ++v) {
      m = fmaxf(m, leaky(acc0[v] * sc + sh));
      m = fmaxf(m, leaky(acc1[v] * sc + sh));
    }
    m = fmaxf(m, __shfl_xor(m, 16, 32));
    if (lane < 16)
      outBase[((size_t)b * NPTS + n) * outStride + o] = (_Float16)m;
  }
}

// ---------------------------------------------------------------------------
// W5 GEMM (16K x 512 x 1024) fused with BN + leaky + max/mean pooling.
// A-operand tiles are streamed with the Tensor Data Mover in a per-wave
// double-buffered pipeline: issue slab c+1 -> s_wait_tensorcnt 1 -> compute
// slab c from LDS. Each wave owns its own LDS slots, so no cross-wave
// barriers in the pipeline; TENSORcnt is per-wave.
// grid = (64 o-tiles, B), block = 256 (8 waves).
// ---------------------------------------------------------------------------
__global__ __launch_bounds__(256) void w5_pool_kernel(
    const _Float16* __restrict__ cat, const _Float16* __restrict__ w5h,
    const float* __restrict__ bn5, float* __restrict__ hmax,
    float* __restrict__ hmean) {
  __shared__ _Float16 abuf[8][2][16 * 128];  // 64 KB, [wave][buf][row*128+c]
  __shared__ float rm[8][16];
  __shared__ float rs[8][16];

  const int ot   = blockIdx.x;
  const int b    = blockIdx.y;
  const int tid  = threadIdx.x;
  const int lane = tid & 31;
  const int wave = tid >> 5;
  const int hi   = lane >> 4;
  const int o    = ot * 16 + (lane & 15);

  const float g  = bn5[o];
  const float be = bn5[1024 + o];
  const float mu = bn5[2048 + o];
  const float va = bn5[3072 + o];
  const float sc = g * rsqrtf(va + EPS);
  const float sh = be - mu * sc;

  const _Float16* bcol  = w5h + (size_t)o * 512;
  const _Float16* aBase = cat + (size_t)b * NPTS * 512;
  __builtin_prefetch(bcol, 0, 1);

  // chunk c (0..31): nt = wave + (c>>2)*8 (M-tile), slab = c&3 (128-half K slab)
  float lmax = -3.0e38f, lsum = 0.f;
  const v8f vzero = {};
  v8f acc = vzero;

  {
    const _Float16* src0 = aBase + (size_t)(wave * 16) * 512;
    stage_chunk(&abuf[wave][0][0], src0, lane);
  }
  for (int c = 0; c < 32; ++c) {
    const int buf = c & 1;
    if (c + 1 < 32) {
      const int nt1   = wave + ((c + 1) >> 2) * 8;
      const int slab1 = (c + 1) & 3;
      const _Float16* src =
          aBase + (size_t)(nt1 * 16) * 512 + slab1 * 128;
      stage_chunk(&abuf[wave][buf ^ 1][0], src, lane);
      stage_wait_le1();
    } else {
      stage_wait_0();
    }

    const int slab = c & 3;
    if (slab == 0) acc = vzero;
    const _Float16* slabp = &abuf[wave][buf][(lane & 15) * 128];
#pragma unroll
    for (int t = 0; t < 4; ++t) {
      v16h A  = load_frag_a(slabp + t * 32, hi);
      v16h Bv = load_frag_b(bcol + slab * 128 + t * 32, hi);
      acc = wmma_f16(A, Bv, acc);
    }
    if (slab == 3) {
#pragma unroll
      for (int v = 0; v < 8; ++v) {
        float h = leaky(acc[v] * sc + sh);
        lmax = fmaxf(lmax, h);
        lsum += h;
      }
    }
  }

  lmax = fmaxf(lmax, __shfl_xor(lmax, 16, 32));
  lsum += __shfl_xor(lsum, 16, 32);
  if (lane < 16) { rm[wave][lane] = lmax; rs[wave][lane] = lsum; }
  __syncthreads();
  if (tid < 16) {
    float m = rm[0][tid], s = rs[0][tid];
#pragma unroll
    for (int w = 1; w < 8; ++w) { m = fmaxf(m, rm[w][tid]); s += rs[w][tid]; }
    hmax[(size_t)b * 1024 + ot * 16 + tid]  = m;
    hmean[(size_t)b * 1024 + ot * 16 + tid] = s * (1.f / (float)NPTS);
  }
}

// ---------------------------------------------------------------------------
// HOSVD head. U3 never formed: X(3) is 1024x48 so core[:,:,r>=48]==0 and
// X(3)^T U3 = V*Sigma from the 48x48 Gram eigenbasis. Jacobi on lane 0.
// ---------------------------------------------------------------------------
__device__ void jacobi_eig(float* A, float* V, int n) {
  for (int i = 0; i < n * n; ++i) V[i] = 0.f;
  for (int i = 0; i < n; ++i) V[i * n + i] = 1.f;
  for (int sweep = 0; sweep < 14; ++sweep) {
    float off = 0.f;
    for (int p = 0; p < n - 1; ++p)
      for (int q = p + 1; q < n; ++q) off += fabsf(A[p * n + q]);
    if (off < 1e-10f) break;
    for (int p = 0; p < n - 1; ++p) {
      for (int q = p + 1; q < n; ++q) {
        float apq = A[p * n + q];
        if (fabsf(apq) < 1e-12f) continue;
        float app = A[p * n + p], aqq = A[q * n + q];
        float tau = (aqq - app) / (2.f * apq);
        float t = (tau >= 0.f ? 1.f : -1.f) /
                  (fabsf(tau) + sqrtf(1.f + tau * tau));
        float c = rsqrtf(1.f + t * t);
        float s = t * c;
        for (int r = 0; r < n; ++r) {
          float arp = A[r * n + p], arq = A[r * n + q];
          A[r * n + p] = c * arp - s * arq;
          A[r * n + q] = s * arp + c * arq;
        }
        for (int r = 0; r < n; ++r) {
          float apr = A[p * n + r], aqr = A[q * n + r];
          A[p * n + r] = c * apr - s * aqr;
          A[q * n + r] = s * apr + c * aqr;
        }
        for (int r = 0; r < n; ++r) {
          float vrp = V[r * n + p], vrq = V[r * n + q];
          V[r * n + p] = c * vrp - s * vrq;
          V[r * n + q] = s * vrp + c * vrq;
        }
      }
    }
  }
}

__device__ void sort_desc(float* A, float* V, float* ev, int n) {
  for (int i = 0; i < n; ++i) ev[i] = A[i * n + i];
  for (int i = 0; i < n; ++i) {
    int best = i;
    for (int j = i + 1; j < n; ++j)
      if (ev[j] > ev[best]) best = j;
    if (best != i) {
      float t = ev[i]; ev[i] = ev[best]; ev[best] = t;
      for (int r = 0; r < n; ++r) {
        float v = V[r * n + i];
        V[r * n + i] = V[r * n + best];
        V[r * n + best] = v;
      }
    }
  }
}

__global__ __launch_bounds__(256) void hosvd_kernel(
    const float* __restrict__ x, const float* __restrict__ Wc,
    const float* __restrict__ bc, float* __restrict__ xx) {
  __shared__ float G1[256], V1[256];
  __shared__ float G2[9], V2[9];
  __shared__ float G3[48 * 48], V3[48 * 48];
  __shared__ float core[16 * 3 * 48];
  __shared__ float evbuf[48];

  const int tid = threadIdx.x;

  {
    int i = tid >> 4, j = tid & 15;
    const float* xi = x + (size_t)i * 3072;
    const float* xj = x + (size_t)j * 3072;
    float s = 0.f;
    for (int f = 0; f < 3072; ++f) s += xi[f] * xj[f];
    G1[tid] = s;
  }
  if (tid < 9) {
    int q1 = tid / 3, q2 = tid % 3;
    float s = 0.f;
    for (int i = 0; i < 16; ++i) {
      const float* a = x + ((size_t)i * 3 + q1) * NPTS;
      const float* b = x + ((size_t)i * 3 + q2) * NPTS;
      for (int k = 0; k < NPTS; ++k) s += a[k] * b[k];
    }
    G2[tid] = s;
  }
  for (int e = tid; e < 48 * 48; e += 256) {
    int a = e / 48, b2 = e % 48;
    const float* pa = x + (size_t)a * NPTS;
    const float* pb = x + (size_t)b2 * NPTS;
    float s = 0.f;
    for (int k = 0; k < NPTS; ++k) s += pa[k] * pb[k];
    G3[e] = s;
  }
  __syncthreads();

  if (tid == 0) {
    jacobi_eig(G1, V1, 16); sort_desc(G1, V1, evbuf, 16);
    jacobi_eig(G2, V2, 3);  sort_desc(G2, V2, evbuf, 3);
    jacobi_eig(G3, V3, 48); sort_desc(G3, V3, evbuf, 48);  // evbuf = lambda3
  }
  __syncthreads();

  for (int e = tid; e < 16 * 3 * 48; e += 256) {
    int p = e / 144, rem = e % 144;
    int q = rem / 48, r = rem % 48;
    float sg = sqrtf(fmaxf(evbuf[r], 0.f));
    float s = 0.f;
    for (int i = 0; i < 16; ++i)
      for (int j = 0; j < 3; ++j)
        s += V1[i * 16 + p] * V2[j * 3 + q] * V3[(i * 3 + j) * 48 + r];
    core[e] = s * sg;
  }
  __syncthreads();

  for (int e = tid; e < 16 * 128; e += 256) {
    int p = e / 128, c = e % 128;
    float s = bc[c];
    for (int q = 0; q < 3; ++q)
      for (int r = 0; r < 48; ++r)
        s += core[(p * 3 + q) * 48 + r] * Wc[(size_t)c * 3072 + q * 1024 + r];
    xx[e] = s;
  }
}

// ---------------------------------------------------------------------------
// Head MLP: feats(2176) -> 512 -> 256 -> 40.
// ---------------------------------------------------------------------------
__global__ __launch_bounds__(256) void head_kernel(
    const float* __restrict__ hmax, const float* __restrict__ hmean,
    const float* __restrict__ xx, const float* __restrict__ L1,
    const float* __restrict__ bn6, const float* __restrict__ L2w,
    const float* __restrict__ L2b, const float* __restrict__ bn7,
    const float* __restrict__ L3w, const float* __restrict__ L3b,
    float* __restrict__ out) {
  __shared__ float feats[2176];
  __shared__ float h1[512];
  __shared__ float h2[256];

  const int b   = blockIdx.x;
  const int tid = threadIdx.x;

  for (int i = tid; i < 1024; i += 256) {
    feats[i]        = hmax[(size_t)b * 1024 + i];
    feats[1024 + i] = hmean[(size_t)b * 1024 + i];
  }
  if (tid < 128) feats[2048 + tid] = xx[(size_t)b * 128 + tid];
  __syncthreads();

  for (int o = tid; o < 512; o += 256) {
    const float* w = L1 + (size_t)o * 2176;
    float s = 0.f;
    for (int f = 0; f < 2176; ++f) s += w[f] * feats[f];
    float sc = bn6[o] * rsqrtf(bn6[3 * 512 + o] + EPS);
    float sh = bn6[512 + o] - bn6[2 * 512 + o] * sc;
    h1[o] = leaky(s * sc + sh);
  }
  __syncthreads();

  {
    const int o = tid;
    const float* w = L2w + (size_t)o * 512;
    float s = L2b[o];
    for (int f = 0; f < 512; ++f) s += w[f] * h1[f];
    float sc = bn7[o] * rsqrtf(bn7[3 * 256 + o] + EPS);
    float sh = bn7[256 + o] - bn7[2 * 256 + o] * sc;
    h2[o] = leaky(s * sc + sh);
  }
  __syncthreads();

  if (tid < 40) {
    const float* w = L3w + (size_t)tid * 256;
    float s = L3b[tid];
    for (int f = 0; f < 256; ++f) s += w[f] * h2[f];
    out[(size_t)b * 40 + tid] = s;
  }
}

// ---------------------------------------------------------------------------
extern "C" void kernel_launch(void* const* d_in, const int* in_sizes, int n_in,
                              void* d_out, int out_size, void* d_ws,
                              size_t ws_size, hipStream_t stream) {
  (void)in_sizes; (void)n_in; (void)out_size; (void)ws_size;

  const float* x   = (const float*)d_in[0];
  const float* W1  = (const float*)d_in[1];
  const float* bn1 = (const float*)d_in[2];
  const float* W2  = (const float*)d_in[3];
  const float* bn2 = (const float*)d_in[4];
  const float* W3  = (const float*)d_in[5];
  const float* bn3 = (const float*)d_in[6];
  const float* W4  = (const float*)d_in[7];
  const float* bn4 = (const float*)d_in[8];
  const float* W5  = (const float*)d_in[9];
  const float* bn5 = (const float*)d_in[10];
  const float* L1  = (const float*)d_in[11];
  const float* bn6 = (const float*)d_in[12];
  const float* L2w = (const float*)d_in[13];
  const float* L2b = (const float*)d_in[14];
  const float* bn7 = (const float*)d_in[15];
  const float* L3w = (const float*)d_in[16];
  const float* L3b = (const float*)d_in[17];
  const float* Wc  = (const float*)d_in[18];
  const float* bc  = (const float*)d_in[19];

  char* ws = (char*)d_ws;
  auto alloc = [&](size_t bytes) -> void* {
    void* p = ws;
    ws += (bytes + 255) & ~(size_t)255;
    return p;
  };
  _Float16* xt   = (_Float16*)alloc((size_t)BATCH * NPTS * 32 * 2);
  _Float16* cat  = (_Float16*)alloc((size_t)BATCH * NPTS * 512 * 2);
  _Float16* W1h  = (_Float16*)alloc((size_t)64 * 32 * 2);
  _Float16* W2h  = (_Float16*)alloc((size_t)64 * 128 * 2);
  _Float16* W3h  = (_Float16*)alloc((size_t)128 * 128 * 2);
  _Float16* W4h  = (_Float16*)alloc((size_t)256 * 256 * 2);
  _Float16* W5h  = (_Float16*)alloc((size_t)1024 * 512 * 2);
  int*      idx  = (int*)alloc((size_t)BATCH * NPTS * KNN * 4);
  float*    hmax = (float*)alloc((size_t)BATCH * 1024 * 4);
  float*    hmn  = (float*)alloc((size_t)BATCH * 1024 * 4);
  float*    xx   = (float*)alloc((size_t)BATCH * 128 * 4);

  prep_xt_kernel<<<(BATCH * NPTS * 32 + 255) / 256, 256, 0, stream>>>(x, xt);
  cvt_w_kernel<<<(64 * 32 + 255) / 256, 256, 0, stream>>>(W1, W1h, 64, 6, 32);
  cvt_w_kernel<<<(64 * 128 + 255) / 256, 256, 0, stream>>>(W2, W2h, 64, 128, 128);
  cvt_w_kernel<<<(128 * 128 + 255) / 256, 256, 0, stream>>>(W3, W3h, 128, 128, 128);
  cvt_w_kernel<<<(256 * 256 + 255) / 256, 256, 0, stream>>>(W4, W4h, 256, 256, 256);
  cvt_w_kernel<<<(1024 * 512 + 255) / 256, 256, 0, stream>>>(W5, W5h, 1024, 512, 512);

  dim3 gK(NPTS / 16, BATCH), gC(NPTS, BATCH);

  knn_kernel<32><<<gK, 256, 0, stream>>>(xt, 32, idx);
  edgeconv_kernel<3, 32, 64><<<gC, 256, 0, stream>>>(xt, 32, idx, bn1, W1h,
                                                     cat + 0, 512);
  knn_kernel<64><<<gK, 256, 0, stream>>>(cat + 0, 512, idx);
  edgeconv_kernel<64, 128, 64><<<gC, 256, 0, stream>>>(cat + 0, 512, idx, bn2,
                                                       W2h, cat + 64, 512);
  knn_kernel<128><<<gK, 256, 0, stream>>>(cat + 64, 512, idx);
  edgeconv_kernel<128, 256, 128><<<gC, 256, 0, stream>>>(cat + 64, 512, idx,
                                                         bn3, W3h, cat + 128,
                                                         512);
  knn_kernel<256><<<gK, 256, 0, stream>>>(cat + 128, 512, idx);
  edgeconv_kernel<256, 512, 256><<<gC, 256, 0, stream>>>(cat + 128, 512, idx,
                                                         bn4, W4h, cat + 256,
                                                         512);

  w5_pool_kernel<<<dim3(64, BATCH), 256, 0, stream>>>(cat, W5h, bn5, hmax, hmn);

  hosvd_kernel<<<1, 256, 0, stream>>>(x, Wc, bc, xx);

  head_kernel<<<BATCH, 256, 0, stream>>>(hmax, hmn, xx, L1, bn6, L2w, L2b, bn7,
                                         L3w, L3b, (float*)d_out);
}
